// LongShortAttention_89678917140634
// MI455X (gfx1250) — matrix operand order repaired and checked by reading, not compiled, not run on hardware
//
#include <hip/hip_runtime.h>
#include <cstddef>

// ---------------------------------------------------------------------------
// Problem constants (from reference): b=2, n=4096, dim=1024, H=16, d=64,
// W=128 (32 windows), S=16 (256 segments), R=1.
// ---------------------------------------------------------------------------
#define Bb    2
#define Nn    4096
#define DIM   1024
#define NH    16
#define DH    64
#define WW    128
#define NSEG  256
#define NWIN  32
#define BHN   32
#define LN_EPS 1e-5f

typedef __attribute__((ext_vector_type(2)))  __bf16 v2bf;
typedef __attribute__((ext_vector_type(8)))  __bf16 v8bf;
typedef __attribute__((ext_vector_type(16))) __bf16 v16bf;
typedef __attribute__((ext_vector_type(8)))  float  v8f;

// ---------------------------------------------------------------------------
// CDNA5 WMMA fragment loaders (wave32 layouts, CDNA5 ISA §7.12.2).
// All LDS row strides are multiples of 8 bf16 -> every load is an aligned
// 16-byte ds_load_b128 (2 per fragment), no scalar u16 packing.
//
// A 16x32 bf16: lane row m = lane&15, half = lane>>4; elements are two runs
//   of 8 contiguous k: [k0+half*8 .. +7] and [k0+16+half*8 .. +7].
// B 32x16 bf16: lane col n = lane&15; contraction c runs [k0+half*8 .. +7] and
//   [k0+16+half*8 .. +7] (same as A) when B is stored n-major ([n][c]).
// C/D 16x16 f32: elem v <-> row m = 8*half + v, col n = lane&15.
// ---------------------------------------------------------------------------
__device__ __forceinline__ v16bf load_a_frag(const __bf16* base, int ld, int m0,
                                             int k0, int lane) {
  const int row  = m0 + (lane & 15);
  const int half = lane >> 4;
  const __bf16* p = base + row * ld + k0 + half * 8;
  v8bf lo = *(const v8bf*)p;
  v8bf hi = *(const v8bf*)(p + 16);
  return __builtin_shufflevector(lo, hi, 0, 1, 2, 3, 4, 5, 6, 7,
                                 8, 9, 10, 11, 12, 13, 14, 15);
}

// B[c][n] with n-major storage: elem = base[n*ld + c]; c-run contiguous.
__device__ __forceinline__ v16bf load_b_nmajor(const __bf16* base, int ld,
                                               int k0, int n0, int lane) {
  const int n    = n0 + (lane & 15);
  const int half = lane >> 4;
  const __bf16* p = base + n * ld + k0 + half * 8;
  v8bf lo = *(const v8bf*)p;
  v8bf hi = *(const v8bf*)(p + 16);
  return __builtin_shufflevector(lo, hi, 0, 1, 2, 3, 4, 5, 6, 7,
                                 8, 9, 10, 11, 12, 13, 14, 15);
}

__device__ __forceinline__ v8f wmma_bf16(v16bf a, v16bf b, v8f c) {
  return __builtin_amdgcn_wmma_f32_16x16x32_bf16(false, a, false, b,
                                                 (short)0, c, false, false);
}

// ---------------------------------------------------------------------------
// Dense GEMM  Out[M,N] = X[M,K] * W[K,N] + bias, fp32 in/out, bf16 WMMA core.
// mode 0: plain row-major output.  mode 1: head-split output (bh, pos, d)
// with scale (q = (xWq+bq)*d^-0.5).
// Tile 128x128, K-step 32, 256 threads = 8 waves, each wave 32x64.
// W tile stored transposed (n-major) in LDS for b128 fragment loads.
// ---------------------------------------------------------------------------
#define GTM 128
#define GTN 128
#define GTK 32
#define GLD (GTK + 8)   // 40 elements -> 80B row stride (16B aligned)

__global__ __launch_bounds__(256) void gemm_bf16_kernel(
    const float* __restrict__ X, const float* __restrict__ Wt,
    const float* __restrict__ bias, float* __restrict__ Out,
    int M, int N, int K, int mode, float scale) {
  __shared__ __bf16 xa[GTM][GLD];    // [m][k]
  __shared__ __bf16 wbT[GTN][GLD];   // [n][k]  (transposed weight tile)

  const int tid  = threadIdx.x;
  const int lane = tid & 31;
  const int wid  = tid >> 5;
  const int wm   = (wid & 3) * 32;
  const int wn   = (wid >> 2) * 64;
  const int bm   = blockIdx.x * GTM;
  const int bn   = blockIdx.y * GTN;

  v8f acc[2][4];
#pragma unroll
  for (int mi = 0; mi < 2; ++mi)
#pragma unroll
    for (int ni = 0; ni < 4; ++ni)
#pragma unroll
      for (int v = 0; v < 8; ++v) acc[mi][ni][v] = 0.0f;

  for (int k0 = 0; k0 < K; k0 += GTK) {
    // stage X tile 128x32 -> bf16 LDS, packed pair stores
#pragma unroll
    for (int i = tid; i < GTM * GTK / 2; i += 256) {
      int r = i >> 4, c = (i & 15) * 2;
      float2 f = *(const float2*)&X[(size_t)(bm + r) * K + k0 + c];
      v2bf p;
      p[0] = (__bf16)f.x;
      p[1] = (__bf16)f.y;
      *(v2bf*)&xa[r][c] = p;
    }
    // stage W tile 32x128 transposed -> wbT[n][k], packed pair stores
#pragma unroll
    for (int i = tid; i < GTK * GTN / 2; i += 256) {
      int c = i >> 4, r = (i & 15) * 2;  // c = n col, r = k row (even)
      v2bf p;
      p[0] = (__bf16)Wt[(size_t)(k0 + r) * N + bn + c];
      p[1] = (__bf16)Wt[(size_t)(k0 + r + 1) * N + bn + c];
      *(v2bf*)&wbT[c][r] = p;
    }
    // prefetch next K tile of X (emits global_prefetch_b8)
    if (k0 + GTK < K)
      __builtin_prefetch(&X[(size_t)(bm + (tid >> 1)) * K + k0 + GTK], 0, 0);
    __syncthreads();

    v16bf afr[2];
    afr[0] = load_a_frag(&xa[0][0], GLD, wm, 0, lane);
    afr[1] = load_a_frag(&xa[0][0], GLD, wm + 16, 0, lane);
    v16bf bfr[4];
#pragma unroll
    for (int ni = 0; ni < 4; ++ni)
      bfr[ni] = load_b_nmajor(&wbT[0][0], GLD, 0, wn + 16 * ni, lane);
#pragma unroll
    for (int mi = 0; mi < 2; ++mi)
#pragma unroll
      for (int ni = 0; ni < 4; ++ni)
        acc[mi][ni] = wmma_bf16(afr[mi], bfr[ni], acc[mi][ni]);
    __syncthreads();
  }

  const int half = lane >> 4;
  const int ncol = lane & 15;
#pragma unroll
  for (int mi = 0; mi < 2; ++mi)
#pragma unroll
    for (int ni = 0; ni < 4; ++ni)
#pragma unroll
      for (int v = 0; v < 8; ++v) {
        int gm = bm + wm + mi * 16 + 8 * half + v;
        int gn = bn + wn + ni * 16 + ncol;
        float val = (acc[mi][ni][v] + bias[gn]) * scale;
        if (mode == 0) {
          Out[(size_t)gm * N + gn] = val;
        } else {  // head-split (b*H + h, pos, d)
          int bi = gm >> 12, pos = gm & (Nn - 1);
          int h = gn >> 6, dd = gn & 63;
          Out[(((size_t)(bi * NH + h)) * Nn + pos) * DH + dd] = val;
        }
      }
}

// ---------------------------------------------------------------------------
// Global branch compression: per (bh, seg): p = softmax(kv_seg @ Wp + bp)
// over S=16, gkv = sum_s p[s]*kv[s], then LayerNorm over d=64.
// ---------------------------------------------------------------------------
__global__ __launch_bounds__(64) void compress_kernel(
    const float* __restrict__ kv, const float* __restrict__ Wp,
    const float* __restrict__ bp, const float* __restrict__ gg,
    const float* __restrict__ gb, float* __restrict__ gkv) {
  const int seg = blockIdx.x;
  const int bh  = blockIdx.y;
  const int tid = threadIdx.x;  // 0..63 = dim
  __shared__ float wgt[16];
  __shared__ float vec[64];

  const float* base = kv + ((size_t)bh * Nn + seg * 16) * DH;

  if (tid < 16) {
    float acc = bp[0];
#pragma unroll 8
    for (int dd = 0; dd < DH; ++dd) acc += base[tid * DH + dd] * Wp[dd];
    wgt[tid] = acc;
  }
  __syncthreads();
  if (tid == 0) {
    float mx = wgt[0];
#pragma unroll
    for (int s = 1; s < 16; ++s) mx = fmaxf(mx, wgt[s]);
    float sum = 0.0f;
#pragma unroll
    for (int s = 0; s < 16; ++s) { float e = __expf(wgt[s] - mx); wgt[s] = e; sum += e; }
    float inv = 1.0f / sum;
#pragma unroll
    for (int s = 0; s < 16; ++s) wgt[s] *= inv;
  }
  __syncthreads();

  float acc = 0.0f;
#pragma unroll
  for (int s = 0; s < 16; ++s) acc += wgt[s] * base[s * DH + tid];
  vec[tid] = acc;
  __syncthreads();

  float mu = 0.0f;
#pragma unroll 8
  for (int i = 0; i < DH; ++i) mu += vec[i];
  mu *= (1.0f / DH);
  float var = 0.0f;
#pragma unroll 8
  for (int i = 0; i < DH; ++i) { float dlt = vec[i] - mu; var += dlt * dlt; }
  var *= (1.0f / DH);
  float out = (acc - mu) * rsqrtf(var + LN_EPS) * gg[tid] + gb[tid];
  gkv[((size_t)bh * NSEG + seg) * DH + tid] = out;
}

// ---------------------------------------------------------------------------
// Attention: one block per (window, bh). 256 threads = 8 waves, each wave owns
// 16 query rows. Keys = 256 global (gkv) + 256 local look-around (LayerNormed
// kv of windows w-1,w). Flash-style online softmax over 8 chunks of 64 keys;
// V == K. Keys kept in LDS in BOTH orders: k_lds[key][d] feeds q.kT B-frags,
// kT_lds[d][key] feeds P.V B-frags -> all fragment loads are ds_load_b128.
// ---------------------------------------------------------------------------
#define ALD (DH + 8)   // 72 elements -> 144B row stride (16B aligned)

__global__ __launch_bounds__(256) void attention_kernel(
    const float* __restrict__ q, const float* __restrict__ kv,
    const float* __restrict__ gkv, const float* __restrict__ lg,
    const float* __restrict__ lb, float* __restrict__ attn_out) {
  const int w    = blockIdx.x;  // window
  const int bh   = blockIdx.y;
  const int tid  = threadIdx.x;
  const int lane = tid & 31;
  const int wid  = tid >> 5;
  const int wrow = wid * 16;
  const int half = lane >> 4;
  const int ncol = lane & 15;

  __shared__ __bf16 q_lds[WW][ALD];    // [row][d]
  __shared__ __bf16 k_lds[64][ALD];    // chunk: [key][d]
  __shared__ __bf16 kT_lds[DH][ALD];   // chunk: [d][key]
  __shared__ __bf16 p_lds[WW][ALD];    // probs [row][key]

  // stage q (already scaled by d^-0.5 in projection), packed pair stores
  const float* qsrc = q + ((size_t)bh * Nn + (size_t)w * WW) * DH;
#pragma unroll
  for (int i = tid; i < WW * DH / 2; i += 256) {
    int r = i >> 5, c = (i & 31) * 2;
    float2 f = *(const float2*)&qsrc[r * DH + c];
    v2bf p;
    p[0] = (__bf16)f.x;
    p[1] = (__bf16)f.y;
    *(v2bf*)&q_lds[r][c] = p;
  }
  __syncthreads();

  v16bf aq[2];
  aq[0] = load_a_frag(&q_lds[0][0], ALD, wrow, 0, lane);
  aq[1] = load_a_frag(&q_lds[0][0], ALD, wrow, 32, lane);

  v8f oacc[4];
#pragma unroll
  for (int t = 0; t < 4; ++t)
#pragma unroll
    for (int v = 0; v < 8; ++v) oacc[t][v] = 0.0f;
  float rm[8], rl[8];
#pragma unroll
  for (int v = 0; v < 8; ++v) { rm[v] = -1e30f; rl[v] = 0.0f; }

  for (int kc = 0; kc < 8; ++kc) {
    __syncthreads();  // previous chunk's k_lds/kT_lds reads complete
    if (kc < 4) {
      const float* src = gkv + ((size_t)bh * NSEG + kc * 64) * DH;
      for (int i = tid; i < 64 * DH; i += 256) {
        int r = i >> 6, dd = i & 63;
        __bf16 bv = (__bf16)src[i];
        k_lds[r][dd]  = bv;
        kT_lds[dd][r] = bv;
      }
    } else if (tid < 64) {
      // local look-around key row j, LayerNorm over d (pad rows are -1.0)
      const int j = (kc - 4) * 64 + tid;              // 0..255
      const long pos = (long)w * WW - WW + j;         // kv position
      const bool pad = (pos < 0);
      const float* src = kv + ((size_t)bh * Nn + (pos < 0 ? 0 : pos)) * DH;
      float s = 0.0f, ss = 0.0f;
#pragma unroll 8
      for (int dd = 0; dd < DH; ++dd) {
        float vv = pad ? -1.0f : src[dd];
        s += vv; ss += vv * vv;
      }
      float mu  = s * (1.0f / DH);
      float var = ss * (1.0f / DH) - mu * mu;
      float rs  = rsqrtf(var + LN_EPS);
#pragma unroll 8
      for (int dd = 0; dd < DH; ++dd) {
        float vv = pad ? -1.0f : src[dd];
        __bf16 bv = (__bf16)((vv - mu) * rs * lg[dd] + lb[dd]);
        k_lds[tid][dd]  = bv;
        kT_lds[dd][tid] = bv;
      }
    }
    __syncthreads();

    // ---- sim chunk: q(16x64) . K^T -> 16x64 in 4 C tiles ----
    v8f sfr[4];
#pragma unroll
    for (int t = 0; t < 4; ++t) {
#pragma unroll
      for (int v = 0; v < 8; ++v) sfr[t][v] = 0.0f;
#pragma unroll
      for (int ks = 0; ks < 2; ++ks) {
        v16bf bk = load_b_nmajor(&k_lds[0][0], ALD, ks * 32, t * 16, lane);
        sfr[t] = wmma_bf16(aq[ks], bk, sfr[t]);
      }
    }

    // ---- masks ----
#pragma unroll
    for (int t = 0; t < 4; ++t) {
      const int key = t * 16 + ncol;  // within chunk
#pragma unroll
      for (int v = 0; v < 8; ++v) {
        const int qrow = wrow + 8 * half + v;  // row within window
        bool ok;
        if (kc < 4) {
          int seg = kc * 64 + key;
          ok = (w * WW + qrow) >= (seg * 16 + 15);
        } else {
          int j = (kc - 4) * 64 + key;
          ok = (j <= qrow + WW) && !((w == 0) && (j < WW));
        }
        if (!ok) sfr[t][v] = -1e30f;
      }
    }

    // ---- online softmax update (row reductions over 16-lane groups) ----
    float mnew[8], alpha[8];
#pragma unroll
    for (int v = 0; v < 8; ++v) {
      float mx = fmaxf(fmaxf(sfr[0][v], sfr[1][v]), fmaxf(sfr[2][v], sfr[3][v]));
#pragma unroll
      for (int off = 1; off < 16; off <<= 1) mx = fmaxf(mx, __shfl_xor(mx, off, 32));
      mnew[v]  = fmaxf(rm[v], mx);
      alpha[v] = __expf(rm[v] - mnew[v]);
      rm[v]    = mnew[v];
    }
#pragma unroll
    for (int t = 0; t < 4; ++t)
#pragma unroll
      for (int v = 0; v < 8; ++v) sfr[t][v] = __expf(sfr[t][v] - mnew[v]);
#pragma unroll
    for (int v = 0; v < 8; ++v) {
      float ps = sfr[0][v] + sfr[1][v] + sfr[2][v] + sfr[3][v];
#pragma unroll
      for (int off = 1; off < 16; off <<= 1) ps += __shfl_xor(ps, off, 32);
      rl[v] = rl[v] * alpha[v] + ps;
#pragma unroll
      for (int t = 0; t < 4; ++t) oacc[t][v] *= alpha[v];
    }

    // ---- stage probs and accumulate P.V (V rows == K rows) ----
#pragma unroll
    for (int t = 0; t < 4; ++t)
#pragma unroll
      for (int v = 0; v < 8; ++v)
        p_lds[wrow + 8 * half + v][t * 16 + ncol] = (__bf16)sfr[t][v];

#pragma unroll
    for (int ks = 0; ks < 2; ++ks) {
      v16bf ap = load_a_frag(&p_lds[0][0], ALD, wrow, ks * 32, lane);
#pragma unroll
      for (int t = 0; t < 4; ++t) {
        v16bf bv = load_b_nmajor(&kT_lds[0][0], ALD, ks * 32, t * 16, lane);
        oacc[t] = wmma_bf16(ap, bv, oacc[t]);
      }
    }
  }

  // ---- epilogue: normalize, scatter to (b, pos, dim) layout ----
  const int bi = bh >> 4;
  const int h  = bh & 15;
#pragma unroll
  for (int v = 0; v < 8; ++v) {
    float inv = rl[v] > 0.0f ? 1.0f / rl[v] : 0.0f;
    int qrow = wrow + 8 * half + v;
    size_t rowbase = ((size_t)bi * Nn + (size_t)w * WW + qrow) * DIM + h * DH;
#pragma unroll
    for (int t = 0; t < 4; ++t)
      attn_out[rowbase + t * 16 + ncol] = oacc[t][v] * inv;
  }
}

// ---------------------------------------------------------------------------
// Launch: q/kv projections -> compress -> attention -> output projection
// ---------------------------------------------------------------------------
extern "C" void kernel_launch(void* const* d_in, const int* in_sizes, int n_in,
                              void* d_out, int out_size, void* d_ws, size_t ws_size,
                              hipStream_t stream) {
  (void)in_sizes; (void)n_in; (void)out_size; (void)ws_size;
  const float* x    = (const float*)d_in[0];
  const float* Wq   = (const float*)d_in[1];
  const float* bq   = (const float*)d_in[2];
  const float* Wkv  = (const float*)d_in[3];
  const float* bkv  = (const float*)d_in[4];
  const float* Wp   = (const float*)d_in[5];
  const float* bp   = (const float*)d_in[6];
  const float* lnlg = (const float*)d_in[7];
  const float* lnlb = (const float*)d_in[8];
  const float* lngg = (const float*)d_in[9];
  const float* lngb = (const float*)d_in[10];
  const float* Wo   = (const float*)d_in[11];
  const float* bo   = (const float*)d_in[12];
  float* out = (float*)d_out;

  const size_t qkv_elems = (size_t)BHN * Nn * DH;  // 8,388,608 floats
  float* q_ws    = (float*)d_ws;
  float* kv_ws   = q_ws + qkv_elems;
  float* gkv_ws  = kv_ws + qkv_elems;
  float* attn_ws = gkv_ws + (size_t)BHN * NSEG * DH;

  const int M = Bb * Nn;  // 8192
  dim3 ggrid(M / GTM, DIM / GTN);  // 64 x 8

  const float qscale = 0.125f;  // d^-0.5, d=64
  gemm_bf16_kernel<<<ggrid, 256, 0, stream>>>(x, Wq, bq, q_ws, M, DIM, DIM, 1, qscale);
  gemm_bf16_kernel<<<ggrid, 256, 0, stream>>>(x, Wkv, bkv, kv_ws, M, DIM, DIM, 1, 1.0f);

  compress_kernel<<<dim3(NSEG, BHN), 64, 0, stream>>>(kv_ws, Wp, bp, lngg, lngb, gkv_ws);

  attention_kernel<<<dim3(NWIN, BHN), 256, 0, stream>>>(q_ws, kv_ws, gkv_ws,
                                                        lnlg, lnlb, attn_ws);

  gemm_bf16_kernel<<<ggrid, 256, 0, stream>>>(attn_ws, Wo, bo, out, M, DIM, DIM, 0, 1.0f);
}
